// CrossAttention_40578851013109
// MI455X (gfx1250) — matrix operand order repaired and checked by reading, not compiled, not run on hardware
//
#include <hip/hip_runtime.h>

// CrossAttention for MI455X (gfx1250): bf16 WMMA GEMMs + fused softmax attention.
// B=4, LQ=LK=1024, HID=OUT=1024, NH=16, HD=64.
//
// Pipeline:
//   1) cvt kernels: f32 -> bf16 for activations and weights (one pass, ws)
//   2) Q/K/V projection GEMMs (bf16 WMMA, async-LDS double-buffered staging)
//   3) fused attention per (b, h, 16 q-rows): scores WMMA -> mask+softmax in
//      a 16x1024 f32 LDS tile -> attn written once (coalesced) -> context WMMA
//   4) output projection GEMM (f32 result to d_out)

#define BB   4
#define LQ   1024
#define LK   1024
#define HID  1024
#define NH   16
#define HD   64

typedef __attribute__((ext_vector_type(4)))  float   v4f;
typedef __attribute__((ext_vector_type(8)))  float   v8f;
typedef __attribute__((ext_vector_type(8)))  __bf16  v8bf;
typedef __attribute__((ext_vector_type(16))) __bf16  v16bf;

static __device__ __forceinline__ __bf16 f2bf(float f) {
  union { float f; unsigned u; } x; x.f = f;
  unsigned r = x.u + 0x7FFFu + ((x.u >> 16) & 1u);   // round-to-nearest-even
  unsigned short h = (unsigned short)(r >> 16);
  __bf16 b; __builtin_memcpy(&b, &h, sizeof(b));
  return b;
}

// ---- CDNA5 async global->LDS staging (ASYNCcnt path), with safe fallback ----
#if defined(__gfx1250__) && __has_builtin(__builtin_amdgcn_global_load_async_to_lds_b128)
#define USE_ASYNC_LDS 1
#else
#define USE_ASYNC_LDS 0
#endif

#if USE_ASYNC_LDS
// Builtin expects int4 pointers: addrspace(1) src (prints as "__device__"),
// addrspace(3) dst (prints as "__shared__") per hipcc diagnostics.
typedef int i4v __attribute__((__vector_size__(4 * sizeof(int))));
static __device__ __forceinline__ __attribute__((address_space(1))) i4v*
glb_i4(const void* p) {
  return (__attribute__((address_space(1))) i4v*)(size_t)p;
}
// generic LDS pointer low 32 bits == LDS byte offset (ISA 10.2 aperture table)
static __device__ __forceinline__ __attribute__((address_space(3))) i4v*
lds_i4(void* p) {
  return (__attribute__((address_space(3))) i4v*)(unsigned)(size_t)p;
}
#if __has_builtin(__builtin_amdgcn_s_wait_asynccnt)
#define WAIT_ASYNC(n) __builtin_amdgcn_s_wait_asynccnt(n)
#else
#define WAIT_ASYNC(n) asm volatile("s_wait_asynccnt %0" ::"i"(n) : "memory")
#endif
#else
#define WAIT_ASYNC(n) ((void)0)
#endif

// ---------------------------------------------------------------------------
// Elementwise f32 -> bf16 (8 elems/thread, n multiple of 8)
// ---------------------------------------------------------------------------
__global__ __launch_bounds__(256)
void cvt_bf16_kernel(const float* __restrict__ src, __bf16* __restrict__ dst, int n) {
  const int i = (blockIdx.x * 256 + threadIdx.x) * 8;
  if (i < n) {
    v4f a = *(const v4f*)(src + i);
    v4f b = *(const v4f*)(src + i + 4);
    v8bf o;
    for (int e = 0; e < 4; ++e) { o[e] = f2bf(a[e]); o[4 + e] = f2bf(b[e]); }
    *(v8bf*)(dst + i) = o;
  }
}

// ---------------------------------------------------------------------------
// C[M,N] = A[M,K] (bf16) @ W[N,K]^T (bf16) + bias(f32), v_wmma_f32_16x16x32_bf16
// Block: 128 threads (4 waves), tile 64x64, K-step 32, double-buffered LDS with
// async global->LDS copies.
// WMMA operand layouts (CDNA5 ISA 7.12.2):
//   A lane L, elem i: M = L%16, K = (L/16)*8 + i + (i>=8 ? 8 : 0)
//   B lane L, elem i: N = L%16, K = (L/16)*8 + i + (i>=8 ? 8 : 0)
//   C lane L, vgpr g: M = g + (L/16)*8, N = L%16
// ---------------------------------------------------------------------------
#define KSTR 40   // LDS row stride (bf16); 80B keeps 16B alignment, skews banks

__global__ __launch_bounds__(128)
void gemm_bt_kernel(const __bf16* __restrict__ A, const __bf16* __restrict__ W,
                    const float* __restrict__ bias,
                    __bf16* __restrict__ Cb, float* __restrict__ Cf,
                    int M, int N, int K) {
  __shared__ __bf16 At[2][64][KSTR];
  __shared__ __bf16 Bt[2][64][KSTR];

  const int tid  = threadIdx.x;
  const int lane = tid & 31;
  const int wave = tid >> 5;
  const int lm   = lane & 15;
  const int lhi  = lane >> 4;
  const int m0   = blockIdx.y * 64;
  const int n0   = blockIdx.x * 64;
  const int wr0  = (wave >> 1) * 32;
  const int wc0  = (wave & 1) * 32;

  const int r    = tid >> 1;           // 0..63 : staging row
  const int half = (tid & 1) * 16;     // 0/16  : staging k-half

  auto stage = [&](int buf, int kb) {
    const __bf16* ga = A + (size_t)(m0 + r) * K + kb + half;
    const __bf16* gw = W + (size_t)(n0 + r) * K + kb + half;
#if USE_ASYNC_LDS
    __builtin_amdgcn_global_load_async_to_lds_b128(glb_i4(ga),     lds_i4(&At[buf][r][half]),     0, 0);
    __builtin_amdgcn_global_load_async_to_lds_b128(glb_i4(ga + 8), lds_i4(&At[buf][r][half + 8]), 0, 0);
    __builtin_amdgcn_global_load_async_to_lds_b128(glb_i4(gw),     lds_i4(&Bt[buf][r][half]),     0, 0);
    __builtin_amdgcn_global_load_async_to_lds_b128(glb_i4(gw + 8), lds_i4(&Bt[buf][r][half + 8]), 0, 0);
#else
    *(v8bf*)&At[buf][r][half]     = *(const v8bf*)ga;
    *(v8bf*)&At[buf][r][half + 8] = *(const v8bf*)(ga + 8);
    *(v8bf*)&Bt[buf][r][half]     = *(const v8bf*)gw;
    *(v8bf*)&Bt[buf][r][half + 8] = *(const v8bf*)(gw + 8);
#endif
  };

  v8f acc[2][2];
  for (int i = 0; i < 2; ++i)
    for (int j = 0; j < 2; ++j) acc[i][j] = v8f{};

  stage(0, 0);
  int ib = 0;
  for (int kb = 0; kb < K; kb += 32, ib ^= 1) {
    const bool more = (kb + 32) < K;
    if (more) {
      stage(ib ^ 1, kb + 32);   // prefetch next tile (4 async ops in flight)
      WAIT_ASYNC(4);            // older batch (current tile) has landed
    } else {
      WAIT_ASYNC(0);
    }
    __syncthreads();

    v16bf afr[2], bfr[2];
    for (int i = 0; i < 2; ++i) {
      const __bf16* ap = &At[ib][wr0 + i * 16 + lm][lhi * 8];
      v8bf alo = *(const v8bf*)ap;
      v8bf ahi = *(const v8bf*)(ap + 16);
      for (int e = 0; e < 8; ++e) { afr[i][e] = alo[e]; afr[i][8 + e] = ahi[e]; }

      const __bf16* bp = &Bt[ib][wc0 + i * 16 + lm][lhi * 8];
      v8bf blo = *(const v8bf*)bp;
      v8bf bhi = *(const v8bf*)(bp + 16);
      for (int e = 0; e < 8; ++e) { bfr[i][e] = blo[e]; bfr[i][8 + e] = bhi[e]; }
    }
    for (int i = 0; i < 2; ++i)
      for (int j = 0; j < 2; ++j)
        acc[i][j] = __builtin_amdgcn_wmma_f32_16x16x32_bf16(
            false, afr[i], false, bfr[j], (short)0, acc[i][j], false, false);
    __syncthreads();
  }

  for (int i = 0; i < 2; ++i)
    for (int j = 0; j < 2; ++j) {
      const int nn = n0 + wc0 + j * 16 + lm;
      const float bv = bias[nn];
      if (Cb) {
        for (int g = 0; g < 8; ++g) {
          const int mm = m0 + wr0 + i * 16 + g + lhi * 8;
          Cb[(size_t)mm * N + nn] = f2bf(acc[i][j][g] + bv);
        }
      } else {
        for (int g = 0; g < 8; ++g) {
          const int mm = m0 + wr0 + i * 16 + g + lhi * 8;
          Cf[(size_t)mm * N + nn] = acc[i][j][g] + bv;
        }
      }
    }
}

// ---------------------------------------------------------------------------
// Fused attention: one block = (batch b, head h, 16 query rows).
// Phase 1: scores = Q*K^T/8 -> 16x1024 f32 tile in LDS (WMMA)
// Phase 2: mask (coalesced int4) + row softmax in LDS, write attn once
// Phase 3: context = attn * V (WMMA, per-32-k transposed V tile), bf16 out
// LDS: 64KB scores + 5KB Vt + 0.5KB reductions ~ 71KB (< 320KB WGP LDS)
// ---------------------------------------------------------------------------
__global__ __launch_bounds__(128)
void attn_kernel(const __bf16* __restrict__ Qp, const __bf16* __restrict__ Kp,
                 const __bf16* __restrict__ Vp, const int* __restrict__ mask,
                 float* __restrict__ attn_out, __bf16* __restrict__ ctx) {
  __shared__ float  sc[16][1024];
  __shared__ __bf16 Vt[64][KSTR];
  __shared__ float  red[16][8];

  const int tid  = threadIdx.x;
  const int lane = tid & 31;
  const int wave = tid >> 5;
  const int lm   = lane & 15;
  const int lhi  = lane >> 4;
  const int q0   = blockIdx.x * 16;
  const int h    = blockIdx.y;
  const int b    = blockIdx.z;

  const size_t qbase = (size_t)b * LQ * HID + (size_t)h * HD;
  const size_t kbase = (size_t)b * LK * HID + (size_t)h * HD;

  // ---- Phase 1: scores ----
  v16bf aq[2];
  for (int c = 0; c < 2; ++c) {
    const __bf16* ap = Qp + qbase + (size_t)(q0 + lm) * HID + c * 32 + lhi * 8;
    v8bf lo = *(const v8bf*)ap;
    v8bf hi = *(const v8bf*)(ap + 16);
    for (int e = 0; e < 8; ++e) { aq[c][e] = lo[e]; aq[c][8 + e] = hi[e]; }
  }
  for (int t = 0; t < 16; ++t) {           // each wave covers 256 key columns
    const int nc0 = wave * 256 + t * 16;
    v8f acc = v8f{};
    for (int c = 0; c < 2; ++c) {
      const __bf16* bp = Kp + kbase + (size_t)(nc0 + lm) * HID + c * 32 + lhi * 8;
      v8bf lo = *(const v8bf*)bp;
      v8bf hi = *(const v8bf*)(bp + 16);
      v16bf bk;
      for (int e = 0; e < 8; ++e) { bk[e] = lo[e]; bk[8 + e] = hi[e]; }
      acc = __builtin_amdgcn_wmma_f32_16x16x32_bf16(
          false, aq[c], false, bk, (short)0, acc, false, false);
    }
    const int ncol = nc0 + lm;
    for (int g = 0; g < 8; ++g)
      sc[g + lhi * 8][ncol] = acc[g] * 0.125f;   // 1/sqrt(HD=64)
  }
  __syncthreads();

  // ---- Phase 2: mask + softmax (8 threads per row, 128 contiguous cols) ----
  {
    const int row = tid >> 3, seg = tid & 7, base = seg * 128;
    const int* mrow = mask + ((size_t)b * LQ + q0 + row) * LK + base;
    float m = -3.4e38f;
    for (int j = 0; j < 128; j += 4) {
      int4 mv = *(const int4*)(mrow + j);          // coalesced mask read
      float4 s4 = *(float4*)&sc[row][base + j];
      s4.x = mv.x ? s4.x : -1e9f;
      s4.y = mv.y ? s4.y : -1e9f;
      s4.z = mv.z ? s4.z : -1e9f;
      s4.w = mv.w ? s4.w : -1e9f;
      *(float4*)&sc[row][base + j] = s4;
      m = fmaxf(m, fmaxf(fmaxf(s4.x, s4.y), fmaxf(s4.z, s4.w)));
    }
    red[row][seg] = m;
    __syncthreads();
    float rm = red[row][0];
    for (int s = 1; s < 8; ++s) rm = fmaxf(rm, red[row][s]);
    __syncthreads();                               // red reused below
    float s = 0.0f;
    for (int j = 0; j < 128; ++j) {
      const float e = __expf(sc[row][base + j] - rm);
      sc[row][base + j] = e;
      s += e;
    }
    red[row][seg] = s;
    __syncthreads();
    float rs = 0.0f;
    for (int s2 = 0; s2 < 8; ++s2) rs += red[row][s2];
    const float inv = 1.0f / rs;
    float* outp = attn_out + (((size_t)b * NH + h) * LQ + q0 + row) * LK + base;
    for (int j = 0; j < 128; j += 4) {
      float4 v;
      v.x = sc[row][base + j + 0] * inv;
      v.y = sc[row][base + j + 1] * inv;
      v.z = sc[row][base + j + 2] * inv;
      v.w = sc[row][base + j + 3] * inv;
      sc[row][base + j + 0] = v.x; sc[row][base + j + 1] = v.y;
      sc[row][base + j + 2] = v.z; sc[row][base + j + 3] = v.w;
      *(float4*)(outp + j) = v;                    // attn written exactly once
    }
  }
  __syncthreads();

  // ---- Phase 3: context = attn * V (wave w -> HD columns w*16..w*16+15) ----
  v8f cacc = v8f{};
  for (int kb = 0; kb < LK; kb += 32) {
    {
      const int kk   = tid >> 2;          // 0..31 key row within chunk
      const int hseg = (tid & 3) * 16;    // 0/16/32/48
      const __bf16* vp = Vp + kbase + (size_t)(kb + kk) * HID + hseg;
      v8bf lo = *(const v8bf*)vp;
      v8bf hi = *(const v8bf*)(vp + 8);
      for (int i = 0; i < 8; ++i) Vt[hseg + i][kk]     = lo[i];
      for (int i = 0; i < 8; ++i) Vt[hseg + 8 + i][kk] = hi[i];
    }
    __syncthreads();

    v16bf af;
    {
      const float* sp = &sc[lm][kb + lhi * 8];
      v4f f0 = *(const v4f*)(sp +  0);
      v4f f1 = *(const v4f*)(sp +  4);
      v4f f2 = *(const v4f*)(sp + 16);
      v4f f3 = *(const v4f*)(sp + 20);
      for (int e = 0; e < 4; ++e) {
        af[e]      = f2bf(f0[e]); af[4 + e]  = f2bf(f1[e]);
        af[8 + e]  = f2bf(f2[e]); af[12 + e] = f2bf(f3[e]);
      }
    }
    v16bf bf;
    {
      const __bf16* bp = &Vt[wave * 16 + lm][lhi * 8];
      v8bf lo = *(const v8bf*)bp;
      v8bf hi = *(const v8bf*)(bp + 16);
      for (int e = 0; e < 8; ++e) { bf[e] = lo[e]; bf[8 + e] = hi[e]; }
    }
    cacc = __builtin_amdgcn_wmma_f32_16x16x32_bf16(
        false, af, false, bf, (short)0, cacc, false, false);
    __syncthreads();
  }

  for (int g = 0; g < 8; ++g) {
    const int qr = g + lhi * 8;
    const int n  = wave * 16 + lm;
    ctx[((size_t)b * LQ + q0 + qr) * HID + h * HD + n] = f2bf(cacc[g]);
  }
}

// ---------------------------------------------------------------------------
// Workspace (64MB):
//   qb 8MB | kb 8MB | vb 8MB | Wq/Wk/Wv/Wo bf16 2MB each | Qp/Kp/Vp 8MB each
//   | ctx(bf16) 8MB
// ---------------------------------------------------------------------------
extern "C" void kernel_launch(void* const* d_in, const int* in_sizes, int n_in,
                              void* d_out, int out_size, void* d_ws, size_t ws_size,
                              hipStream_t stream) {
  (void)in_sizes; (void)n_in; (void)out_size; (void)ws_size;
  const float* query = (const float*)d_in[0];
  const float* key   = (const float*)d_in[1];
  const float* value = (const float*)d_in[2];
  const int*   mask  = (const int*)d_in[3];
  const float* Wq = (const float*)d_in[4];
  const float* bq = (const float*)d_in[5];
  const float* Wk = (const float*)d_in[6];
  const float* bk = (const float*)d_in[7];
  const float* Wv = (const float*)d_in[8];
  const float* bv = (const float*)d_in[9];
  const float* Wo = (const float*)d_in[10];
  const float* bo = (const float*)d_in[11];

  float* out      = (float*)d_out;
  float* attn_out = out + (size_t)BB * LQ * HID;   // tuple: (output, attn)

  char* ws = (char*)d_ws;
  const size_t MB = (size_t)1 << 20;
  __bf16* qb  = (__bf16*)(ws + 0 * MB);
  __bf16* kb  = (__bf16*)(ws + 8 * MB);
  __bf16* vb  = (__bf16*)(ws + 16 * MB);
  __bf16* wqb = (__bf16*)(ws + 24 * MB);
  __bf16* wkb = (__bf16*)(ws + 26 * MB);
  __bf16* wvb = (__bf16*)(ws + 28 * MB);
  __bf16* wob = (__bf16*)(ws + 30 * MB);
  __bf16* Qp  = (__bf16*)(ws + 32 * MB);
  __bf16* Kp  = (__bf16*)(ws + 40 * MB);
  __bf16* Vp  = (__bf16*)(ws + 48 * MB);
  __bf16* ctx = (__bf16*)(ws + 56 * MB);

  const int nAct = BB * LQ * HID;   // 4M
  const int nWgt = HID * HID;       // 1M
  cvt_bf16_kernel<<<nAct / (256 * 8), 256, 0, stream>>>(query, qb, nAct);
  cvt_bf16_kernel<<<nAct / (256 * 8), 256, 0, stream>>>(key,   kb, nAct);
  cvt_bf16_kernel<<<nAct / (256 * 8), 256, 0, stream>>>(value, vb, nAct);
  cvt_bf16_kernel<<<nWgt / (256 * 8), 256, 0, stream>>>(Wq, wqb, nWgt);
  cvt_bf16_kernel<<<nWgt / (256 * 8), 256, 0, stream>>>(Wk, wkb, nWgt);
  cvt_bf16_kernel<<<nWgt / (256 * 8), 256, 0, stream>>>(Wv, wvb, nWgt);
  cvt_bf16_kernel<<<nWgt / (256 * 8), 256, 0, stream>>>(Wo, wob, nWgt);

  const dim3 gb(128);
  const dim3 gg(HID / 64, (BB * LQ) / 64);   // 16 x 64 blocks

  gemm_bt_kernel<<<gg, gb, 0, stream>>>(qb, wqb, bq, Qp, nullptr, BB * LQ, HID, HID);
  gemm_bt_kernel<<<gg, gb, 0, stream>>>(kb, wkb, bk, Kp, nullptr, BB * LK, HID, HID);
  gemm_bt_kernel<<<gg, gb, 0, stream>>>(vb, wvb, bv, Vp, nullptr, BB * LK, HID, HID);

  attn_kernel<<<dim3(LQ / 16, NH, BB), 128, 0, stream>>>(Qp, Kp, Vp, mask, attn_out, ctx);

  gemm_bt_kernel<<<gg, gb, 0, stream>>>(ctx, wob, bo, nullptr, out, BB * LQ, HID, HID);
}
